// MAB_1_83021717832758
// MI455X (gfx1250) — matrix operand order repaired and checked by reading, not compiled, not run on hardware
//
#include <hip/hip_runtime.h>
#include <hip/hip_bf16.h>

typedef _Float16 v16h __attribute__((ext_vector_type(16)));
typedef _Float16 v8h  __attribute__((ext_vector_type(8)));
typedef float    v8f  __attribute__((ext_vector_type(8)));
typedef unsigned int v4u __attribute__((ext_vector_type(4)));
typedef int          v4i __attribute__((ext_vector_type(4)));
typedef int          v8i __attribute__((ext_vector_type(8)));

#define NB 4
#define ND 512
#define NH 8
#define NDK 64
#define NSQ 2048
#define NSK 2048

// attention scale (dk^-0.5) with log2(e) folded in: softmax runs in base-2
#define QSCALE (0.125f * 1.4426950408889634f)

__device__ __forceinline__ v16h load2x8(const _Float16* p0, const _Float16* p1) {
    v8h a = *(const v8h*)p0;
    v8h b = *(const v8h*)p1;
    v16h r;
#pragma unroll
    for (int i = 0; i < 8; ++i) { r[i] = a[i]; r[8 + i] = b[i]; }
    return r;
}

__device__ __forceinline__ v8f wmma16(v16h a, v16h b, v8f c) {
    return __builtin_amdgcn_wmma_f32_16x16x32_f16(false, a, false, b, (short)0, c, false, false);
}

// ---------------------------------------------------------------------------
// Tensor Data Mover: 2-D tile (f16 elements) global -> LDS, with LDS row
// padding (pad_interval code 4 = every 32 DWORDs, pad_amount code 3 = 4 DWORDs
// => 64-half rows land on a 72-half pitch). D# per CDNA5 ISA 8.3/8.4.
// ---------------------------------------------------------------------------
__device__ __forceinline__ void tdm_load_tile_f16(
    unsigned lds_off, const _Float16* gptr,
    unsigned tensor_d0, unsigned tensor_d1,
    unsigned tile_d0, unsigned tile_d1,
    unsigned long long stride0_elems)
{
    unsigned long long ga = (unsigned long long)(size_t)gptr;
    v4u g0;
    g0[0] = 1u;                                   // count=1, is_restore=0
    g0[1] = lds_off;                              // lds_addr (bytes)
    g0[2] = (unsigned)(ga & 0xffffffffu);         // global_addr[31:0]
    g0[3] = (unsigned)((ga >> 32) & 0x01ffffffu)  // global_addr[56:32]
            | (2u << 30);                         // type = 2 (image)
    v8i g1;
    // data_size=1 (2B), pad_enable, pad_interval=4 (32 DW), pad_amount=3 (4 DW)
    g1[0] = (int)(0x00010000u | (1u << 20) | (4u << 22) | (3u << 25));
    g1[1] = (int)((tensor_d0 & 0xffffu) << 16);               // td0[15:0] @bit48
    g1[2] = (int)(((tensor_d0 >> 16) & 0xffffu) |
                  ((tensor_d1 & 0xffffu) << 16));             // td0 hi | td1 lo
    g1[3] = (int)(((tensor_d1 >> 16) & 0xffffu) |
                  ((tile_d0 & 0xffffu) << 16));               // td1 hi | tile0
    g1[4] = (int)(tile_d1 & 0xffffu);                         // tile1 | tile2=0
    g1[5] = (int)(unsigned)(stride0_elems & 0xffffffffu);     // stride0[31:0]
    g1[6] = (int)(unsigned)((stride0_elems >> 32) & 0xffffu); // stride0 hi | stride1=0
    g1[7] = 0;
    v4i z; z[0] = 0; z[1] = 0; z[2] = 0; z[3] = 0;
#if defined(__clang_major__) && (__clang_major__ >= 23)
    v8i z8; z8[0]=0; z8[1]=0; z8[2]=0; z8[3]=0; z8[4]=0; z8[5]=0; z8[6]=0; z8[7]=0;
    __builtin_amdgcn_tensor_load_to_lds(g0, g1, z, z, z8, 0);
#else
    __builtin_amdgcn_tensor_load_to_lds(g0, g1, z, z, 0);
#endif
}

// ---------------------------------------------------------------------------
__global__ void cvt_f32_f16(const float* __restrict__ src, _Float16* __restrict__ dst, int n) {
    int i = blockIdx.x * 256 + threadIdx.x;
    if (i < n) dst[i] = (_Float16)src[i];
}

// [C=512, S] fp32 -> [S, C=512] f16, tiled LDS transpose
__global__ void transpose_cvt(const float* __restrict__ X, _Float16* __restrict__ XT, int S) {
    __shared__ float tile[32][33];
    int b = blockIdx.z;
    int s0 = blockIdx.x * 32, c0 = blockIdx.y * 32;
    int tx = threadIdx.x, ty = threadIdx.y; // 32 x 8
    const float* xb = X + (size_t)b * ND * S;
#pragma unroll
    for (int i = 0; i < 32; i += 8)
        tile[ty + i][tx] = xb[(size_t)(c0 + ty + i) * S + s0 + tx];
    __syncthreads();
    _Float16* xtb = XT + ((size_t)b * S + s0) * ND + c0;
#pragma unroll
    for (int i = 0; i < 32; i += 8)
        xtb[(size_t)(ty + i) * ND + tx] = (_Float16)tile[tx][ty + i];
}

// ---------------------------------------------------------------------------
// WMMA GEMM: out[o,s] = sum_c W[o,c]*XT[s,c] (+bias). One wave = 16(o) x 64(s)
// macro-tile: A-fragment reused over 4 independent accumulators.
// ---------------------------------------------------------------------------
__global__ __launch_bounds__(128) void gemm_wmma(
    const _Float16* __restrict__ W,   // [512][512] f16
    const _Float16* __restrict__ XT,  // [B][S][512] f16
    const float* __restrict__ bias,   // [512]
    int mode, int S,
    _Float16* __restrict__ out16,
    float* __restrict__ xbuf)
{
    int wave = threadIdx.x >> 5, lane = threadIdx.x & 31;
    int tilesS4 = S >> 6;                 // 64-wide s supertiles
    int perB = 32 * tilesS4;
    int tile = blockIdx.x * 4 + wave;
    int b  = tile / perB;
    int r  = tile % perB;
    int o0 = (r / tilesS4) * 16;
    int s0 = (r % tilesS4) * 64;

    int hl = lane & 15;
    bool hi = lane >= 16;
    int rowOff = hi ? 8 : 0;

    const _Float16* xb = XT + (size_t)b * S * ND;
    v8f acc[4] = {};
#pragma unroll 2
    for (int k = 0; k < ND; k += 32) {
        const _Float16* wr = W + (size_t)(o0 + hl) * ND + k + (hi ? 8 : 0);
        v16h a = load2x8(wr, wr + 16);
#pragma unroll
        for (int t = 0; t < 4; ++t) {
            const _Float16* xr = xb + (size_t)(s0 + t * 16 + hl) * ND + k + (hi ? 8 : 0);
            v16h bm = load2x8(xr, xr + 16);
            acc[t] = wmma16(a, bm, acc[t]);
        }
    }

    int h = o0 >> 6;
    int dk0 = (o0 & 63) + rowOff;
#pragma unroll
    for (int t = 0; t < 4; ++t) {
        int scol = s0 + t * 16 + hl;
        if (mode == 0) {
            size_t qsb = (((size_t)b * NH + h) * S + scol) * NDK + dk0;
            size_t xb0 = ((size_t)b * S + scol) * ND + o0 + rowOff;
#pragma unroll
            for (int j = 0; j < 8; ++j) {
                float val = acc[t][j] + bias[o0 + rowOff + j];
                out16[qsb + j] = (_Float16)(val * QSCALE);  // dk^-0.5 * log2(e)
                xbuf[xb0 + j] = val;                        // residual copy
            }
        } else if (mode == 1) {
            size_t kb = (((size_t)b * NH + h) * S + scol) * NDK + dk0;
#pragma unroll
            for (int j = 0; j < 8; ++j)
                out16[kb + j] = (_Float16)(acc[t][j] + bias[o0 + rowOff + j]);
        } else if (mode == 2) {
            size_t vb = (((size_t)b * NH + h) * NDK + dk0) * (size_t)S + scol;
#pragma unroll
            for (int j = 0; j < 8; ++j)
                out16[vb + (size_t)j * S] = (_Float16)(acc[t][j] + bias[o0 + rowOff + j]);
        } else {
            size_t xb0 = ((size_t)b * S + scol) * ND + o0 + rowOff;
#pragma unroll
            for (int j = 0; j < 8; ++j) {
                float y = acc[t][j] + bias[o0 + rowOff + j];
                xbuf[xb0 + j] += (y > 0.f ? y : 0.f);
            }
        }
    }
}

// ---------------------------------------------------------------------------
// Flash attention. Grid: (SQ/64, B*H). 4 waves x 16 query rows.
// 64-key chunks; K/V tiles double-buffered in LDS via the Tensor Data Mover.
// Softmax runs in base-2 (scale pre-folded with log2 e) -> bare v_exp_f32.
// ---------------------------------------------------------------------------
__global__ __launch_bounds__(128) void attn_kernel(
    const _Float16* __restrict__ qs,  // [B,H,SQ,64] pre-scaled by QSCALE
    const _Float16* __restrict__ kk,  // [B,H,SK,64]
    const _Float16* __restrict__ vT,  // [B,H,64,SK]
    float* __restrict__ xbuf)         // [B,SQ,512] rmw (holds q residual)
{
    __shared__ alignas(16) _Float16 kT[2][64][72];     // [key][dk], 72 pitch (TDM pad)
    __shared__ alignas(16) _Float16 vTile[2][64][72];  // [dk][key]
    __shared__ alignas(16) _Float16 pS[4][16][72];     // per-wave P scratch

    int tid = threadIdx.x, wave = tid >> 5, lane = tid & 31;
    int bh = blockIdx.y;
    int b = bh >> 3, h = bh & 7;
    int q0 = blockIdx.x * 64 + wave * 16;
    int hl = lane & 15;
    bool hiL = lane >= 16;
    int rowOff = hiL ? 8 : 0;

    const _Float16* qsb = qs + (((size_t)bh) * NSQ + q0) * NDK;
    const _Float16* kkb = kk + (size_t)bh * NSK * NDK;
    const _Float16* vtb = vT + (size_t)bh * NDK * NSK;

    // resident Q A-fragments (dk 0..31, 32..63)
    const _Float16* qr = qsb + (size_t)hl * NDK + (hiL ? 8 : 0);
    v16h aq0 = load2x8(qr, qr + 16);
    v16h aq1 = load2x8(qr + 32, qr + 48);

    float m[8], l[8];
    v8f O[4] = {};
#pragma unroll
    for (int j = 0; j < 8; ++j) { m[j] = -1e30f; l[j] = 0.f; }

    const int nChunk = NSK / 64;
    if (wave == 0) {
        tdm_load_tile_f16((unsigned)(size_t)&kT[0][0][0], kkb,
                          NDK, NSK, NDK, 64, NDK);
        tdm_load_tile_f16((unsigned)(size_t)&vTile[0][0][0], vtb,
                          NSK, NDK, 64, NDK, NSK);
    }

    for (int c = 0; c < nChunk; ++c) {
        int cur = c & 1;
        if (wave == 0) {
            if (c + 1 < nChunk) {
                int key1 = (c + 1) * 64;
                tdm_load_tile_f16((unsigned)(size_t)&kT[1 - cur][0][0],
                                  kkb + (size_t)key1 * NDK, NDK, NSK, NDK, 64, NDK);
                tdm_load_tile_f16((unsigned)(size_t)&vTile[1 - cur][0][0],
                                  vtb + key1, NSK, NDK, 64, NDK, NSK);
                __builtin_amdgcn_s_wait_tensorcnt((short)2); // chunk c complete
            } else {
                __builtin_amdgcn_s_wait_tensorcnt((short)0);
            }
        }
        __syncthreads();

        // S2 = (q * scale * log2e) @ K^T : four 16-key tiles, 2 dk-steps each
        v8f sc[4];
#pragma unroll
        for (int t = 0; t < 4; ++t) {
            const _Float16* kr0 = &kT[cur][t * 16 + hl][hiL ? 8 : 0];
            v16h b0 = load2x8(kr0, kr0 + 16);
            v16h b1 = load2x8(kr0 + 32, kr0 + 48);
            v8f cacc = {};
            cacc = wmma16(aq0, b0, cacc);
            cacc = wmma16(aq1, b1, cacc);
            sc[t] = cacc;
        }

        // base-2 online softmax (row lives across a 16-lane half; xor 1,2,4,8)
        float alpha[8];
#pragma unroll
        for (int j = 0; j < 8; ++j) {
            float mx = fmaxf(fmaxf(sc[0][j], sc[1][j]), fmaxf(sc[2][j], sc[3][j]));
            mx = fmaxf(mx, __shfl_xor(mx, 1));
            mx = fmaxf(mx, __shfl_xor(mx, 2));
            mx = fmaxf(mx, __shfl_xor(mx, 4));
            mx = fmaxf(mx, __shfl_xor(mx, 8));
            float mn = fmaxf(m[j], mx);
            alpha[j] = exp2f(m[j] - mn);
            m[j] = mn;
            float rs = 0.f;
#pragma unroll
            for (int t = 0; t < 4; ++t) {
                float p = exp2f(sc[t][j] - mn);
                sc[t][j] = p;
                rs += p;
            }
            rs += __shfl_xor(rs, 1);
            rs += __shfl_xor(rs, 2);
            rs += __shfl_xor(rs, 4);
            rs += __shfl_xor(rs, 8);
            l[j] = l[j] * alpha[j] + rs;
        }
#pragma unroll
        for (int d = 0; d < 4; ++d)
#pragma unroll
            for (int j = 0; j < 8; ++j) O[d][j] *= alpha[j];

        // C-layout P -> LDS -> two A-layout fragments (keys 0..31, 32..63)
#pragma unroll
        for (int j = 0; j < 8; ++j)
#pragma unroll
            for (int t = 0; t < 4; ++t)
                pS[wave][j + rowOff][t * 16 + hl] = (_Float16)sc[t][j];
        const _Float16* pr = &pS[wave][hl][hiL ? 8 : 0];
        v16h pa0 = load2x8(pr, pr + 16);
        v16h pa1 = load2x8(pr + 32, pr + 48);

        // O += P @ V (4 dk-tiles x 2 key-steps)
#pragma unroll
        for (int d = 0; d < 4; ++d) {
            const _Float16* vr = &vTile[cur][d * 16 + hl][hiL ? 8 : 0];
            v16h bv0 = load2x8(vr, vr + 16);
            v16h bv1 = load2x8(vr + 32, vr + 48);
            O[d] = wmma16(pa0, bv0, O[d]);
            O[d] = wmma16(pa1, bv1, O[d]);
        }
        __syncthreads(); // all waves done with buf[cur] before it is refilled
    }

    float inv[8];
#pragma unroll
    for (int j = 0; j < 8; ++j) inv[j] = 1.0f / l[j];
#pragma unroll
    for (int d = 0; d < 4; ++d)
#pragma unroll
        for (int j = 0; j < 8; ++j) {
            size_t idx = ((size_t)b * NSQ + q0 + j + rowOff) * ND + h * NDK + d * 16 + hl;
            xbuf[idx] += O[d][j] * inv[j];
        }
}

// ---------------------------------------------------------------------------
// LN0: over channel axis (contiguous 512 per row), in-place f32 + f16 copy.
// ---------------------------------------------------------------------------
__global__ __launch_bounds__(256) void ln_kernel(
    float* __restrict__ xbuf, _Float16* __restrict__ xT16,
    const float* __restrict__ gamma, const float* __restrict__ beta)
{
    int wave = threadIdx.x >> 5, lane = threadIdx.x & 31;
    int row = blockIdx.x * 8 + wave;
    float* xr = xbuf + (size_t)row * ND;

    float v[16], sum = 0.f, ss = 0.f;
#pragma unroll
    for (int i = 0; i < 16; ++i) {
        v[i] = xr[lane + i * 32];
        sum += v[i];
        ss  += v[i] * v[i];
    }
    for (int msk = 1; msk < 32; msk <<= 1) {
        sum += __shfl_xor(sum, msk);
        ss  += __shfl_xor(ss,  msk);
    }
    float u = sum * (1.f / 512.f);
    float var = ss * (1.f / 512.f) - u * u;
    float rstd = rsqrtf(var + 1e-12f);
#pragma unroll
    for (int i = 0; i < 16; ++i) {
        int c = lane + i * 32;
        float y = gamma[c] * (v[i] - u) * rstd + beta[c];
        xr[c] = y;
        xT16[(size_t)row * ND + c] = (_Float16)y;
    }
}

// ---------------------------------------------------------------------------
// LN1 + transpose: per 32-row group compute row stats, then stream 32x32
// tiles through LDS so the final [B, D, S] store is coalesced in s.
// ---------------------------------------------------------------------------
__global__ __launch_bounds__(256) void ln1t_kernel(
    const float* __restrict__ xbuf, const float* __restrict__ gamma,
    const float* __restrict__ beta, float* __restrict__ outF, int S)
{
    __shared__ float stats_u[32];
    __shared__ float stats_r[32];
    __shared__ float tile[32][33];
    int tid = threadIdx.x;
    int wave = tid >> 5, lane = tid & 31;
    int rowBlk = blockIdx.x;                 // [0, B*S/32)
    int s0 = (rowBlk * 32) % S;
    int b  = (rowBlk * 32) / S;
    const float* xb = xbuf + ((size_t)b * S + s0) * ND;

    // per-row stats: each wave handles 4 rows, coalesced reads
#pragma unroll
    for (int rr = 0; rr < 4; ++rr) {
        int r = wave * 4 + rr;
        const float* xr = xb + (size_t)r * ND;
        float sum = 0.f, ss = 0.f;
#pragma unroll
        for (int i = 0; i < 16; ++i) {
            float x = xr[lane + i * 32];
            sum += x; ss += x * x;
        }
        for (int msk = 1; msk < 32; msk <<= 1) {
            sum += __shfl_xor(sum, msk);
            ss  += __shfl_xor(ss,  msk);
        }
        float u = sum * (1.f / 512.f);
        float var = ss * (1.f / 512.f) - u * u;
        if (lane == 0) { stats_u[r] = u; stats_r[r] = rsqrtf(var + 1e-12f); }
    }
    __syncthreads();

    // normalize + transpose-store 32x32 tiles
    int tx = tid & 31, tw = tid >> 5;        // 32 x 8
    for (int c0 = 0; c0 < ND; c0 += 32) {
#pragma unroll
        for (int i = 0; i < 32; i += 8) {
            int r = tw + i;
            float x = xb[(size_t)r * ND + c0 + tx];
            tile[r][tx] = gamma[c0 + tx] * (x - stats_u[r]) * stats_r[r] + beta[c0 + tx];
        }
        __syncthreads();
#pragma unroll
        for (int i = 0; i < 32; i += 8) {
            int c = tw + i;
            outF[((size_t)b * ND + c0 + c) * S + s0 + tx] = tile[tx][c];
        }
        __syncthreads();
    }
}

// ---------------------------------------------------------------------------
extern "C" void kernel_launch(void* const* d_in, const int* in_sizes, int n_in,
                              void* d_out, int out_size, void* d_ws, size_t ws_size,
                              hipStream_t stream) {
    (void)in_sizes; (void)n_in; (void)out_size; (void)ws_size;
    const float* Q      = (const float*)d_in[0];
    const float* K      = (const float*)d_in[1];
    const float* Wq     = (const float*)d_in[2];
    const float* bq     = (const float*)d_in[3];
    const float* Wk     = (const float*)d_in[4];
    const float* bk     = (const float*)d_in[5];
    const float* Wv     = (const float*)d_in[6];
    const float* bv     = (const float*)d_in[7];
    const float* Wo     = (const float*)d_in[8];
    const float* bo     = (const float*)d_in[9];
    const float* gamma0 = (const float*)d_in[10];
    const float* beta0  = (const float*)d_in[11];
    const float* gamma1 = (const float*)d_in[12];
    const float* beta1  = (const float*)d_in[13];
    float* outF = (float*)d_out;

    char* ws = (char*)d_ws;
    size_t off = 0;
    const size_t WN  = (size_t)ND * ND;
    const size_t SDN = (size_t)NB * NSQ * ND;
    _Float16* Wq16 = (_Float16*)(ws + off); off += WN * 2;
    _Float16* Wk16 = (_Float16*)(ws + off); off += WN * 2;
    _Float16* Wv16 = (_Float16*)(ws + off); off += WN * 2;
    _Float16* Wo16 = (_Float16*)(ws + off); off += WN * 2;
    _Float16* QT   = (_Float16*)(ws + off); off += SDN * 2;
    _Float16* KT   = (_Float16*)(ws + off); off += SDN * 2;
    _Float16* qsB  = (_Float16*)(ws + off); off += SDN * 2;
    _Float16* kkB  = (_Float16*)(ws + off); off += SDN * 2;
    _Float16* vTB  = (_Float16*)(ws + off); off += SDN * 2;
    _Float16* ln0T = (_Float16*)(ws + off); off += SDN * 2;
    float*    xbuf = (float*)(ws + off);    off += SDN * 4;

    cvt_f32_f16<<<(int)(WN / 256), 256, 0, stream>>>(Wq, Wq16, (int)WN);
    cvt_f32_f16<<<(int)(WN / 256), 256, 0, stream>>>(Wk, Wk16, (int)WN);
    cvt_f32_f16<<<(int)(WN / 256), 256, 0, stream>>>(Wv, Wv16, (int)WN);
    cvt_f32_f16<<<(int)(WN / 256), 256, 0, stream>>>(Wo, Wo16, (int)WN);

    transpose_cvt<<<dim3(NSQ / 32, ND / 32, NB), dim3(32, 8), 0, stream>>>(Q, QT, NSQ);
    transpose_cvt<<<dim3(NSK / 32, ND / 32, NB), dim3(32, 8), 0, stream>>>(K, KT, NSK);

    gemm_wmma<<<1024, 128, 0, stream>>>(Wq16, QT, bq, 0, NSQ, qsB, xbuf);
    gemm_wmma<<<1024, 128, 0, stream>>>(Wk16, KT, bk, 1, NSK, kkB, xbuf);
    gemm_wmma<<<1024, 128, 0, stream>>>(Wv16, KT, bv, 2, NSK, vTB, xbuf);

    attn_kernel<<<dim3(NSQ / 64, NB * NH), 128, 0, stream>>>(qsB, kkB, vTB, xbuf);

    ln_kernel<<<(NB * NSQ) / 8, 256, 0, stream>>>(xbuf, ln0T, gamma0, beta0);
    gemm_wmma<<<1024, 128, 0, stream>>>(Wo16, ln0T, bo, 3, NSQ, nullptr, xbuf);
    ln1t_kernel<<<(NB * NSQ) / 32, 256, 0, stream>>>(xbuf, gamma1, beta1, outF, NSQ);
}
